// GVAE_73555609911878
// MI455X (gfx1250) — compile-verified, matrix operand order checked by reading, not run on hardware
//
#include <hip/hip_runtime.h>
#include <hip/hip_bf16.h>
#include <stdint.h>

// ---------------------------------------------------------------------------
// GVAE forward for MI455X (gfx1250, wave32, WMMA).
// All dense GEMMs use v_wmma_f32_16x16x32_f16; SpMM uses L2-resident f32
// atomics (feature matrix + accumulator both fit the 192MB L2). z@z.T
// (51.2 GFLOP, 1.6 GB output) is the dominant stage: write-bandwidth-bound
// (~69us at 23.3 TB/s) on the WMMA path, so its stores are non-temporal to
// keep the streaming output from rinsing the L2-resident z operand.
// ---------------------------------------------------------------------------

typedef __attribute__((ext_vector_type(16))) _Float16 v16h;
typedef __attribute__((ext_vector_type(8)))  float    v8f;

#define NN 20000
#define NE 640000

// ------------------------- elementwise helpers -----------------------------

__global__ void zero_f32(float* __restrict__ p, long long n) {
    long long i = (long long)blockIdx.x * blockDim.x + threadIdx.x;
    if (i < n) p[i] = 0.0f;
}

__global__ void cvt_f32_f16(const float* __restrict__ in, _Float16* __restrict__ out,
                            long long n) {
    long long i = (long long)blockIdx.x * blockDim.x + threadIdx.x;
    if (i < n) out[i] = (_Float16)in[i];
}

__global__ void relu_cvt_f32_f16(const float* __restrict__ in, _Float16* __restrict__ out,
                                 long long n) {
    long long i = (long long)blockIdx.x * blockDim.x + threadIdx.x;
    if (i < n) {
        float v = in[i];
        out[i] = (_Float16)(v > 0.0f ? v : 0.0f);
    }
}

// in[K,N] f32 row-major -> out[N,K] f16 row-major (i.e. B^T staging)
__global__ void transpose_cvt(const float* __restrict__ in, _Float16* __restrict__ out,
                              int K, int N) {
    long long i = (long long)blockIdx.x * blockDim.x + threadIdx.x;
    if (i < (long long)K * N) {
        int k = (int)(i / N);
        int n = (int)(i % N);
        out[(long long)n * K + k] = (_Float16)in[i];
    }
}

// ------------------------------- SpMM --------------------------------------
// One block per edge; lane d handles feature d. Gathers and atomics both hit
// L2 (arrays are ~20MB), so this is on-chip-bandwidth bound, not HBM bound.

__global__ void spmm_edge(const int* __restrict__ src, const int* __restrict__ dst,
                          const float* __restrict__ w, const float* __restrict__ feat,
                          float* __restrict__ out, int D) {
    int e = blockIdx.x;
    int d = threadIdx.x;
    float v = w[e] * feat[(long long)src[e] * D + d];
    atomicAdd(&out[(long long)dst[e] * D + d], v);
}

// Feature width 128 = [mu(64) | logvar(64)]; scatter into the two output
// regions of d_out directly.
__global__ void spmm_edge_split(const int* __restrict__ src, const int* __restrict__ dst,
                                const float* __restrict__ w, const float* __restrict__ feat,
                                float* __restrict__ mu_out, float* __restrict__ lv_out) {
    int e = blockIdx.x;
    int d = threadIdx.x;                 // 0..127
    float v = w[e] * feat[(long long)src[e] * 128 + d];
    long long r = (long long)dst[e] * 64;
    if (d < 64) atomicAdd(&mu_out[r + d], v);
    else        atomicAdd(&lv_out[r + d - 64], v);
}

// ------------------------------ WMMA GEMM ----------------------------------
// C[M,N] f32 = A[M,K] f16 (row-major) * Bt[N,K] f16 (row-major B-transpose).
// 256 threads = 8 waves; block tile 128(M) x 64(N); each wave owns 32x32.
//
// 16-bit A 16x32 fragment layout (ISA 7.12.2): lane<16 (M=lane) holds K=0..7
// in v0..3 and K=16..23 in v4..7; lane>=16 holds K=8..15 and K=24..31.
// That is exactly two contiguous 16B chunks at half-offset base and base+16,
// base = kstep*32 + (lane&16 ? 8 : 0). B fragments (Bt row = column of B)
// load with the identical pattern.
//
// STREAM_OUT=true emits non-temporal stores (CDNA5 store TH=NT) for
// write-once streaming outputs (the 1.6GB adjacency matrix).

__device__ __forceinline__ v16h load_frag16(const _Float16* __restrict__ base,
                                            int row, int ld, int kb, int lane) {
    const _Float16* p = base + (long long)row * ld + kb + ((lane & 16) ? 8 : 0);
    union { v16h h; uint4 u[2]; } f;
    f.u[0] = *(const uint4*)(p);        // halves base   .. base+7
    f.u[1] = *(const uint4*)(p + 16);   // halves base+16.. base+23
    return f.h;
}

template <bool STREAM_OUT>
__device__ __forceinline__ void store_tile(float* __restrict__ C, v8f c,
                                           int rbase, int col, int M, int N, int ldc) {
    if (col >= N) return;
#pragma unroll
    for (int i = 0; i < 8; ++i) {
        int row = rbase + i;
        if (row < M) {
            float* p = C + (long long)row * ldc + col;
            if (STREAM_OUT) __builtin_nontemporal_store(c[i], p);
            else            *p = c[i];
        }
    }
}

template <bool STREAM_OUT>
__global__ void wmma_gemm_f16(const _Float16* __restrict__ A,
                              const _Float16* __restrict__ Bt,
                              float* __restrict__ C,
                              int M, int N, int K) {
    const int lane = threadIdx.x & 31;
    const int wave = threadIdx.x >> 5;
    const int m0 = blockIdx.y * 128 + (wave >> 1) * 32;   // wave's 32-row strip
    const int n0 = blockIdx.x * 64  + (wave & 1)  * 32;   // wave's 32-col strip
    const int l15 = lane & 15;

    // Clamp fragment rows; out-of-range results are masked at store time.
    const int ar0 = min(m0      + l15, M - 1);
    const int ar1 = min(m0 + 16 + l15, M - 1);
    const int br0 = min(n0      + l15, N - 1);
    const int br1 = min(n0 + 16 + l15, N - 1);

    v8f c00 = {}, c01 = {}, c10 = {}, c11 = {};

    for (int kb = 0; kb < K; kb += 32) {
        v16h a0 = load_frag16(A,  ar0, K, kb, lane);
        v16h a1 = load_frag16(A,  ar1, K, kb, lane);
        v16h b0 = load_frag16(Bt, br0, K, kb, lane);
        v16h b1 = load_frag16(Bt, br1, K, kb, lane);
        c00 = __builtin_amdgcn_wmma_f32_16x16x32_f16(false, a0, false, b0, (short)0, c00, false, false);
        c01 = __builtin_amdgcn_wmma_f32_16x16x32_f16(false, a0, false, b1, (short)0, c01, false, false);
        c10 = __builtin_amdgcn_wmma_f32_16x16x32_f16(false, a1, false, b0, (short)0, c10, false, false);
        c11 = __builtin_amdgcn_wmma_f32_16x16x32_f16(false, a1, false, b1, (short)0, c11, false, false);
    }

    // C/D layout: lane<16 -> M=0..7 over v0..7, lane>=16 -> M=8..15; N = lane&15.
    const int rb = ((lane & 16) ? 8 : 0);
    store_tile<STREAM_OUT>(C, c00, m0      + rb, n0      + l15, M, N, N);
    store_tile<STREAM_OUT>(C, c01, m0      + rb, n0 + 16 + l15, M, N, N);
    store_tile<STREAM_OUT>(C, c10, m0 + 16 + rb, n0      + l15, M, N, N);
    store_tile<STREAM_OUT>(C, c11, m0 + 16 + rb, n0 + 16 + l15, M, N, N);
}

// ------------------------------- driver ------------------------------------

extern "C" void kernel_launch(void* const* d_in, const int* in_sizes, int n_in,
                              void* d_out, int out_size, void* d_ws, size_t ws_size,
                              hipStream_t stream) {
    const float* x    = (const float*)d_in[0];   // [20000,512]
    const int*   esrc = (const int*)  d_in[1];   // [E]
    const int*   edst = (const int*)  d_in[2];   // [E]
    const float* ew   = (const float*)d_in[3];   // [E]
    const float* W1   = (const float*)d_in[4];   // [512,256]
    const float* W2   = (const float*)d_in[5];   // [256,64]
    const float* W3   = (const float*)d_in[6];   // [256,64]

    float* adj    = (float*)d_out;                       // [20000,20000]
    float* mu_out = (float*)d_out + 400000000LL;         // [20000,64]
    float* lv_out = (float*)d_out + 401280000LL;         // [20000,64]

    char* ws = (char*)d_ws;                              // ~84.8 MB used
    float*    h0   = (float*)   (ws + 0);                // [20000,256] f32
    float*    agg1 = (float*)   (ws + 20480000);         // [20000,256] f32
    _Float16* xh   = (_Float16*)(ws + 40960000);         // [20000,512] f16
    _Float16* h1h  = (_Float16*)(ws + 61440000);         // [20000,256] f16
    float*    mlp  = (float*)   (ws + 71680000);         // [20000,128] f32 (mu|lv pre-agg)
    _Float16* w1t  = (_Float16*)(ws + 81920000);         // [256,512]  f16 = W1^T
    _Float16* w23t = (_Float16*)(ws + 82182144);         // [128,256]  f16 = [W2^T;W3^T]
    _Float16* zh   = (_Float16*)(ws + 82247680);         // [20000,64] f16

    // Zero accumulators (d_out / d_ws arrive poisoned).
    zero_f32<<<(5120000 + 255) / 256, 256, 0, stream>>>(agg1, 5120000LL);
    zero_f32<<<(2560000 + 255) / 256, 256, 0, stream>>>(mu_out, 2560000LL); // mu+lv contiguous

    // Stage f16 operands.
    cvt_f32_f16<<<(10240000 + 255) / 256, 256, 0, stream>>>(x, xh, 10240000LL);
    transpose_cvt<<<(131072 + 255) / 256, 256, 0, stream>>>(W1, w1t, 512, 256);
    transpose_cvt<<<(16384 + 255) / 256, 256, 0, stream>>>(W2, w23t, 256, 64);
    transpose_cvt<<<(16384 + 255) / 256, 256, 0, stream>>>(W3, w23t + 64 * 256, 256, 64);

    // h0 = x @ W1   [20000,512]x[512,256]  (h0 re-read by spmm -> cached stores)
    {
        dim3 grid(256 / 64, (NN + 127) / 128);
        wmma_gemm_f16<false><<<grid, 256, 0, stream>>>(xh, w1t, h0, NN, 256, 512);
    }

    // agg1 = spmm(adj, h0)
    spmm_edge<<<NE, 256, 0, stream>>>(esrc, edst, ew, h0, agg1, 256);

    // h1 = relu(agg1) -> f16
    relu_cvt_f32_f16<<<(5120000 + 255) / 256, 256, 0, stream>>>(agg1, h1h, 5120000LL);

    // [mu_pre | lv_pre] = h1 @ [W2 | W3]   [20000,256]x[256,128]
    {
        dim3 grid(128 / 64, (NN + 127) / 128);
        wmma_gemm_f16<false><<<grid, 256, 0, stream>>>(h1h, w23t, mlp, NN, 128, 256);
    }

    // mu = spmm(adj, mu_pre); logvar = spmm(adj, lv_pre)
    spmm_edge_split<<<NE, 128, 0, stream>>>(esrc, edst, ew, mlp, mu_out, lv_out);

    // z = mu -> f16
    cvt_f32_f16<<<(1280000 + 255) / 256, 256, 0, stream>>>(mu_out, zh, 1280000LL);

    // adj_rec = z @ z.T   [20000,64]x[64,20000]; 1.6GB write-once output ->
    // non-temporal stores so the stream doesn't evict zh from L2.
    {
        dim3 grid((NN + 63) / 64, (NN + 127) / 128);
        wmma_gemm_f16<true><<<grid, 256, 0, stream>>>(zh, zh, adj, NN, NN, 64);
    }
}
